// LSTMClassifier_32908039422153
// MI455X (gfx1250) — compile-verified, matrix operand order checked by reading, not run on hardware
//
#include <hip/hip_runtime.h>
#include <hip/hip_bf16.h>
#include <math.h>

// ---------------------------------------------------------------------------
// LSTM text classifier: emb lookup -> x@W+b -> 256-step LSTM scan (h@U) ->
// dense head.  bf16 WMMA (v_wmma_f32_16x16x32_bf16) for both GEMMs, f32 state.
// Round 2: 2x2 register blocking per wave (4 wmma per K-chunk, 1.0 frag/wmma).
// ---------------------------------------------------------------------------

typedef __attribute__((ext_vector_type(16))) __bf16 v16bf;
typedef __attribute__((ext_vector_type(8)))  __bf16 v8bf;
typedef __attribute__((ext_vector_type(8)))  float  v8f;

static constexpr int EMBD  = 512;
static constexpr int HID   = 1024;
static constexpr int G4H   = 4096;   // 4*HID, gate order i,f,g,o
static constexpr int BATCH = 32;
static constexpr int SEQ   = 256;
static constexpr int ROWS  = SEQ * BATCH;  // 8192 rows of XA/XZ, r = s*32+b

union FragBF { v16bf v; v8bf h[2]; };

// ---------------------------------------------------------------------------
// Kernel 1: convert W,U -> transposed bf16 (column-major: Bt[col][k]) and
// zero-init h0 (bf16) and c (f32).
// ---------------------------------------------------------------------------
__global__ void prep_kernel(const float* __restrict__ W, const float* __restrict__ U,
                            __bf16* __restrict__ Wtr, __bf16* __restrict__ Utr,
                            __bf16* __restrict__ hbf0, float* __restrict__ c) {
  const long NW = (long)EMBD * G4H;   // 2,097,152
  const long NU = (long)HID * G4H;    // 4,194,304
  const long NH = (long)BATCH * HID;  // 32,768
  const long total = NW + NU + NH;
  for (long i = (long)blockIdx.x * blockDim.x + threadIdx.x; i < total;
       i += (long)gridDim.x * blockDim.x) {
    if (i < NW) {
      long k = i / G4H, j = i % G4H;
      Wtr[j * EMBD + k] = (__bf16)W[i];
    } else if (i < NW + NU) {
      long t = i - NW;
      long k = t / G4H, j = t % G4H;
      Utr[j * HID + k] = (__bf16)U[t];
    } else {
      long t = i - NW - NU;
      hbf0[t] = (__bf16)0.0f;
      c[t] = 0.0f;
    }
  }
}

// ---------------------------------------------------------------------------
// Kernel 2: embedding gather, f32 -> bf16.  XA row r = s*32 + b.
// ---------------------------------------------------------------------------
__global__ void embed_kernel(const int* __restrict__ tokens, const float* __restrict__ emb,
                             __bf16* __restrict__ XA) {
  const long total = (long)ROWS * EMBD;
  for (long i = (long)blockIdx.x * blockDim.x + threadIdx.x; i < total;
       i += (long)gridDim.x * blockDim.x) {
    long r = i >> 9;           // /512
    int  e = (int)(i & 511);
    int  s = (int)(r / BATCH);
    int  b = (int)(r % BATCH);
    int  tok = tokens[b * SEQ + s];   // tokens[B,S]
    XA[i] = (__bf16)emb[(long)tok * EMBD + e];
  }
}

// ---------------------------------------------------------------------------
// Kernel 3: XZ = XA @ W + bias.  M=8192, N=4096, K=512.
// WG = 256 threads = 8 waves, tile 64x128.  Waves arranged 2 (M) x 4 (N);
// each wave computes a 32x32 macro-tile = 2x2 subtiles (4 accumulators).
// Per K-chunk: 2 A frags + 2 B frags -> 4 wmma.
// ---------------------------------------------------------------------------
__global__ __launch_bounds__(256)
void xw_gemm_kernel(const __bf16* __restrict__ XA, const __bf16* __restrict__ Wtr,
                    const float* __restrict__ bias, float* __restrict__ XZ) {
  const int wave = threadIdx.x >> 5;
  const int lane = threadIdx.x & 31;
  const int lm   = lane & 15;
  const int halfk = (lane >> 4) * 8;     // A/B K-half select per lane group
  const int wr = wave & 1;               // M wave-block (0..1)
  const int wc = wave >> 1;              // N wave-block (0..3)

  const int tileM = blockIdx.y * 64;
  const int tileN = blockIdx.x * 128;

  const int M0 = tileM + wr * 32 + lm;        // A-fragment rows (2 subtiles)
  const int M1 = M0 + 16;
  const int n0 = tileN + wc * 32 + lm;        // output columns (2 subtiles)
  const int n1 = n0 + 16;

  v8f acc00, acc01, acc10, acc11;             // [Msub][Nsub]
  const float bv0 = bias[n0];
  const float bv1 = bias[n1];
#pragma unroll
  for (int r = 0; r < 8; ++r) {
    acc00[r] = bv0; acc01[r] = bv1;
    acc10[r] = bv0; acc11[r] = bv1;
  }

  const __bf16* Ar0 = XA  + (long)M0 * EMBD;
  const __bf16* Ar1 = XA  + (long)M1 * EMBD;
  const __bf16* Bc0 = Wtr + (long)n0 * EMBD;
  const __bf16* Bc1 = Wtr + (long)n1 * EMBD;

#pragma unroll 4
  for (int kc = 0; kc < EMBD; kc += 32) {
    const int kb = kc + halfk;
    FragBF a0, a1, b0, b1;
    a0.h[0] = *(const v8bf*)(Ar0 + kb);
    a0.h[1] = *(const v8bf*)(Ar0 + kb + 16);
    a1.h[0] = *(const v8bf*)(Ar1 + kb);
    a1.h[1] = *(const v8bf*)(Ar1 + kb + 16);
    b0.h[0] = *(const v8bf*)(Bc0 + kb);
    b0.h[1] = *(const v8bf*)(Bc0 + kb + 16);
    b1.h[0] = *(const v8bf*)(Bc1 + kb);
    b1.h[1] = *(const v8bf*)(Bc1 + kb + 16);
    acc00 = __builtin_amdgcn_wmma_f32_16x16x32_bf16(false, a0.v, false, b0.v,
                                                    (short)0, acc00, false, false);
    acc01 = __builtin_amdgcn_wmma_f32_16x16x32_bf16(false, a0.v, false, b1.v,
                                                    (short)0, acc01, false, false);
    acc10 = __builtin_amdgcn_wmma_f32_16x16x32_bf16(false, a1.v, false, b0.v,
                                                    (short)0, acc10, false, false);
    acc11 = __builtin_amdgcn_wmma_f32_16x16x32_bf16(false, a1.v, false, b1.v,
                                                    (short)0, acc11, false, false);
  }

  // C/D layout: VGPR r, lane: m = (lane>>4)*8 + r, n = lane&15
  const int mr0 = tileM + wr * 32 + (lane >> 4) * 8;
  const int mr1 = mr0 + 16;
#pragma unroll
  for (int r = 0; r < 8; ++r) {
    XZ[(long)(mr0 + r) * G4H + n0] = acc00[r];
    XZ[(long)(mr0 + r) * G4H + n1] = acc01[r];
    XZ[(long)(mr1 + r) * G4H + n0] = acc10[r];
    XZ[(long)(mr1 + r) * G4H + n1] = acc11[r];
  }
}

// ---------------------------------------------------------------------------
// Kernel 4 (launched SEQ times): one LSTM step.
// 64 workgroups x 64 threads (2 waves).  Wave w owns gates {2w, 2w+1}; block
// owns hidden columns [n0, n0+16).  Per K-chunk: 2 A frags (h rows 0-15 /
// 16-31, reused across both gates) + 2 B frags -> 4 wmma.  Gates combined
// through LDS, writes c, h(f32), h(bf16 ping-pong).
// ---------------------------------------------------------------------------
__global__ __launch_bounds__(64)
void lstm_step_kernel(const float* __restrict__ XZs, const __bf16* __restrict__ hbf_in,
                      const __bf16* __restrict__ Utr, float* __restrict__ c,
                      float* __restrict__ h, __bf16* __restrict__ hbf_out) {
  __shared__ float zLDS[4][32][16];

  const int wave = threadIdx.x >> 5;   // 0 -> gates {i,f}, 1 -> gates {g,o}
  const int lane = threadIdx.x & 31;
  const int lm   = lane & 15;
  const int halfk = (lane >> 4) * 8;
  const int n0 = blockIdx.x * 16;
  const int g0 = wave * 2;
  const int g1 = g0 + 1;
  const int j0 = g0 * HID + n0 + lm;   // columns in [0,4096)
  const int j1 = g1 * HID + n0 + lm;

  // accumulators initialized from xz (contains bias + x@W)
  const int m0 = (lane >> 4) * 8;
  v8f accA0, accA1, accB0, accB1;      // [gate][Msub]
#pragma unroll
  for (int r = 0; r < 8; ++r) {
    accA0[r] = XZs[(long)(m0 + r) * G4H + j0];
    accA1[r] = XZs[(long)(m0 + 16 + r) * G4H + j0];
    accB0[r] = XZs[(long)(m0 + r) * G4H + j1];
    accB1[r] = XZs[(long)(m0 + 16 + r) * G4H + j1];
  }

  const __bf16* A0 = hbf_in + (long)lm * HID;         // batch rows 0..15
  const __bf16* A1 = hbf_in + (long)(lm + 16) * HID;  // batch rows 16..31
  const __bf16* Bc0 = Utr + (long)j0 * HID;           // U columns (transposed)
  const __bf16* Bc1 = Utr + (long)j1 * HID;

#pragma unroll 4
  for (int kc = 0; kc < HID; kc += 32) {
    const int kb = kc + halfk;
    FragBF a0, a1, b0, b1;
    a0.h[0] = *(const v8bf*)(A0 + kb);
    a0.h[1] = *(const v8bf*)(A0 + kb + 16);
    a1.h[0] = *(const v8bf*)(A1 + kb);
    a1.h[1] = *(const v8bf*)(A1 + kb + 16);
    b0.h[0] = *(const v8bf*)(Bc0 + kb);
    b0.h[1] = *(const v8bf*)(Bc0 + kb + 16);
    b1.h[0] = *(const v8bf*)(Bc1 + kb);
    b1.h[1] = *(const v8bf*)(Bc1 + kb + 16);
    accA0 = __builtin_amdgcn_wmma_f32_16x16x32_bf16(false, a0.v, false, b0.v,
                                                    (short)0, accA0, false, false);
    accA1 = __builtin_amdgcn_wmma_f32_16x16x32_bf16(false, a1.v, false, b0.v,
                                                    (short)0, accA1, false, false);
    accB0 = __builtin_amdgcn_wmma_f32_16x16x32_bf16(false, a0.v, false, b1.v,
                                                    (short)0, accB0, false, false);
    accB1 = __builtin_amdgcn_wmma_f32_16x16x32_bf16(false, a1.v, false, b1.v,
                                                    (short)0, accB1, false, false);
  }

  // spill z tiles to LDS for cross-wave (cross-gate) combine
#pragma unroll
  for (int r = 0; r < 8; ++r) {
    zLDS[g0][m0 + r][lm]      = accA0[r];
    zLDS[g0][m0 + 16 + r][lm] = accA1[r];
    zLDS[g1][m0 + r][lm]      = accB0[r];
    zLDS[g1][m0 + 16 + r][lm] = accB1[r];
  }
  __syncthreads();

  for (int t = threadIdx.x; t < BATCH * 16; t += 64) {
    const int m = t >> 4;        // batch row
    const int n = t & 15;        // hidden col within block
    const float zi = zLDS[0][m][n];
    const float zf = zLDS[1][m][n];
    const float zg = zLDS[2][m][n];
    const float zo = zLDS[3][m][n];
    const float ig = 1.0f / (1.0f + __expf(-zi));
    const float fg = 1.0f / (1.0f + __expf(-zf));
    const float gg = tanhf(zg);
    const float og = 1.0f / (1.0f + __expf(-zo));
    const long ci = (long)m * HID + n0 + n;
    const float cn = fg * c[ci] + ig * gg;
    const float hn = og * tanhf(cn);
    c[ci] = cn;
    h[ci] = hn;
    hbf_out[ci] = (__bf16)hn;
  }
}

// ---------------------------------------------------------------------------
// Kernel 5: logits = h @ Wd + bd   (32x5, K=1024) -- tiny, scalar.
// ---------------------------------------------------------------------------
__global__ void dense_kernel(const float* __restrict__ h, const float* __restrict__ Wd,
                             const float* __restrict__ bd, float* __restrict__ out) {
  const int t = threadIdx.x;
  if (t < BATCH * 5) {
    const int b = t / 5, l = t % 5;
    float s = bd[l];
    for (int k = 0; k < HID; ++k) s += h[b * HID + k] * Wd[k * 5 + l];
    out[t] = s;
  }
}

// ---------------------------------------------------------------------------
extern "C" void kernel_launch(void* const* d_in, const int* in_sizes, int n_in,
                              void* d_out, int out_size, void* d_ws, size_t ws_size,
                              hipStream_t stream) {
  const int*   tokens = (const int*)d_in[0];
  const float* emb    = (const float*)d_in[1];
  const float* W      = (const float*)d_in[2];
  const float* U      = (const float*)d_in[3];
  const float* bias   = (const float*)d_in[4];
  const float* Wd     = (const float*)d_in[5];
  const float* bd     = (const float*)d_in[6];
  float* out = (float*)d_out;

  char* ws = (char*)d_ws;
  size_t off = 0;
  auto alloc = [&](size_t bytes) -> void* {
    void* p = ws + off;
    off += (bytes + 255) & ~(size_t)255;
    return p;
  };
  __bf16* Wtr  = (__bf16*)alloc((size_t)EMBD * G4H * 2);   //   4 MB
  __bf16* Utr  = (__bf16*)alloc((size_t)HID  * G4H * 2);   //   8 MB
  __bf16* XA   = (__bf16*)alloc((size_t)ROWS * EMBD * 2);  //   8 MB
  float*  XZ   = (float*)alloc((size_t)ROWS * G4H * 4);    // 128 MB
  __bf16* hbf0 = (__bf16*)alloc((size_t)BATCH * HID * 2);
  __bf16* hbf1 = (__bf16*)alloc((size_t)BATCH * HID * 2);
  float*  cst  = (float*)alloc((size_t)BATCH * HID * 4);
  float*  hst  = (float*)alloc((size_t)BATCH * HID * 4);

  prep_kernel<<<2048, 256, 0, stream>>>(W, U, Wtr, Utr, hbf0, cst);
  embed_kernel<<<4096, 256, 0, stream>>>(tokens, emb, XA);

  dim3 g1(G4H / 128, ROWS / 64);   // 32 x 128 tiles of 128x64
  xw_gemm_kernel<<<g1, 256, 0, stream>>>(XA, Wtr, bias, XZ);

  for (int s = 0; s < SEQ; ++s) {
    const __bf16* hin  = (s & 1) ? hbf1 : hbf0;
    __bf16*       hout = (s & 1) ? hbf0 : hbf1;
    lstm_step_kernel<<<HID / 16, 64, 0, stream>>>(XZ + (long)s * BATCH * G4H,
                                                  hin, Utr, cst, hst, hout);
  }

  dense_kernel<<<1, 256, 0, stream>>>(hst, Wd, bd, out);
}
